// LDPCBeliefPropagation_14001593385499
// MI455X (gfx1250) — compile-verified, matrix operand order checked by reading, not run on hardware
//
#include <hip/hip_runtime.h>
#include <hip/hip_bf16.h>

typedef __attribute__((ext_vector_type(16))) _Float16 v16h;
typedef __attribute__((ext_vector_type(8)))  float    v8f;

#define C_DIM 256   // check nodes (rows of H)
#define V_DIM 512   // variable nodes (cols of H)

// ---------------- workspace layout (bytes) ----------------
//  m_vtoc f32 [V][C] : 0          .. 524288
//  m_ctov f32 [C][V] : 524288     .. 1048576
//  HT     f16 [V][C] : 1048576    .. 1310720   (HT[v][c] = H[c][v])
//  Hmask  u32 [C][16]: 1310720    .. 1327104   (bit v%32 of word v/32 = H[c][v])

__global__ void ldpc_init_kernel(const int* __restrict__ H,
                                 float* __restrict__ m_vtoc,
                                 float* __restrict__ m_ctov,
                                 _Float16* __restrict__ HT,
                                 unsigned* __restrict__ Hmask) {
    int idx = blockIdx.x * blockDim.x + threadIdx.x;
    int stride = gridDim.x * blockDim.x;
    for (int i = idx; i < V_DIM * C_DIM; i += stride) m_vtoc[i] = 1.0f;
    for (int i = idx; i < C_DIM * V_DIM; i += stride) m_ctov[i] = 0.0f;
    for (int i = idx; i < V_DIM * C_DIM; i += stride) {
        int v = i / C_DIM, c = i % C_DIM;
        HT[i] = (_Float16)(float)H[c * V_DIM + v];
    }
    for (int i = idx; i < C_DIM * 16; i += stride) {
        int c = i / 16, w = i % 16;
        unsigned m = 0;
        for (int b = 0; b < 32; ++b)
            if (H[c * V_DIM + w * 32 + b] != 0) m |= (1u << b);
        Hmask[i] = m;
    }
}

__device__ __forceinline__ float phi_f(float s) {
    return 2.0f * atanf(expf(0.5f * s));
}
__device__ __forceinline__ float sgn_f(float x) {
    return x > 0.0f ? 1.0f : (x < 0.0f ? -1.0f : 0.0f);
}

__global__ void __launch_bounds__(256, 1)
ldpc_bp_kernel(const float* __restrict__ llr_g,
               const int* __restrict__ max_iter_p,
               float* __restrict__ m_vtoc,
               float* __restrict__ m_ctov,
               const _Float16* __restrict__ HT,
               const unsigned* __restrict__ Hmask,
               int* __restrict__ out) {
    __shared__ float llr[V_DIM];
    __shared__ float col_f32[C_DIM];
    __shared__ __align__(32) _Float16 col_f16[C_DIM];
    __shared__ __align__(32) _Float16 zero_f16[C_DIM];  // stays all-zero
    __shared__ float base_s[V_DIM];   // raw col@Hf (mv subtracted by consumer)
    __shared__ float mv_s[V_DIM];     // m_vtoc[:, i] prefetched per step
    __shared__ unsigned maskrow_s[16];// H row i bitmask prefetched per step
    __shared__ float red[C_DIM];

    const int tid  = threadIdx.x;
    const int lane = tid & 31;
    const int wave = tid >> 5;

    llr[tid]       = llr_g[tid];
    llr[tid + 256] = llr_g[tid + 256];
    zero_f16[tid]  = (_Float16)0.0f;

    // Thread tid owns check column c = tid in phase 1; cache its H row bitmask.
    unsigned mrow[16];
    #pragma unroll
    for (int w = 0; w < 16; ++w) mrow[w] = Hmask[tid * 16 + w];

    // A-operand source: lanes holding A row M=0 (lane%16==0) read the live col
    // vector; all other lanes read a permanently-zero LDS region. One pointer
    // select for the whole kernel replaces per-WMMA element masking.
    const int  khalf = (lane >> 4) * 16;                    // K sub-block: 0 or 16
    const _Float16* abase = ((lane & 15) == 0) ? col_f16 : zero_f16;

    const int max_iter = *max_iter_p;
    __syncthreads();

    for (int it = 0; it < max_iter; ++it) {
        // ---------- Phase 1: variable-to-check sweep (serial over i=0..V-1) ----
        // Each thread keeps P[c] = prod_{v : H[c,v]=1} tanh(0.5*m_vtoc[v,c])
        // and updates it incrementally -> O(1) per serial step.
        float P = 1.0f;
        for (int w = 0; w < 16; ++w) {
            unsigned m = mrow[w];
            while (m) {
                int b = __ffs((int)m) - 1;
                m &= m - 1u;
                int v = w * 32 + b;
                P *= tanhf(0.5f * m_vtoc[v * C_DIM + tid]);
            }
        }
        for (int i = 0; i < V_DIM; ++i) {
            float oldv = m_vtoc[i * C_DIM + tid];
            float val  = sgn_f(llr[i]) * P;
            m_vtoc[i * C_DIM + tid] = val;
            if ((mrow[i >> 5] >> (i & 31)) & 1u)
                P = P * tanhf(0.5f * val) / tanhf(0.5f * oldv);
        }
        __syncthreads();

        // ---------- Phase 2: check-to-variable sweep (serial over i=0..C-1) ----
        for (int i = 0; i < C_DIM; ++i) {
            // Step prologue (all coalesced / parallel, overlaps the barrier):
            //   col  = m_ctov[:, i]  (rows < i already updated this sweep)
            //   mv_s = m_vtoc[:, i]
            //   maskrow_s = H[i, :] bitmask
            float cv = m_ctov[tid * V_DIM + i];
            col_f32[tid] = cv;
            col_f16[tid] = (_Float16)cv;
            mv_s[tid]       = m_vtoc[(size_t)tid * C_DIM + i];
            mv_s[tid + 256] = m_vtoc[(size_t)(tid + 256) * C_DIM + i];
            if (tid < 16) maskrow_s[tid] = Hmask[i * 16 + tid];
            __syncthreads();

            // Load the 8 A-chunks once; identical for all j-tiles of this wave.
            v16h a8[8];
            #pragma unroll
            for (int kc = 0; kc < 8; ++kc)
                a8[kc] = *(const v16h*)(abase + kc * 32 + khalf);

            // base = col @ Hf via WMMA GEMV (8 waves x 4 tiles = 512 outputs).
            // B tiles are loop-invariant over i -> compiler keeps them resident.
            #pragma unroll
            for (int t = 0; t < 4; ++t) {
                const int j0 = (wave + t * 8) * 16;
                v8f acc = {};
                #pragma unroll
                for (int kc = 0; kc < 8; ++kc) {
                    v16h b = *(const v16h*)(HT + (size_t)(j0 + (lane & 15)) * C_DIM
                                               + kc * 32 + khalf);
                    acc = __builtin_amdgcn_wmma_f32_16x16x32_f16(
                              false, a8[kc], false, b, (short)0, acc, false, false);
                }
                if (lane < 16) base_s[j0 + lane] = acc[0];   // raw dot product
            }
            __syncthreads();

            const float vnew = phi_f(base_s[i] - mv_s[i]);
            const float coli = col_f32[i];
            #pragma unroll
            for (int r = 0; r < 2; ++r) {
                int j = tid + r * 256;
                unsigned wbits = maskrow_s[j >> 5];
                float hf  = (float)((wbits >> (j & 31)) & 1u);
                float adj = (j > i) ? hf * (vnew - coli) : 0.0f;
                m_ctov[i * V_DIM + j] = phi_f(base_s[j] - mv_s[j] + adj);
            }
            __syncthreads();
        }
    }

    // ---------- final hard decision for outputs 0..3 ----------
    for (int v = 0; v < 4; ++v) {
        red[tid] = tanhf(0.5f * m_ctov[tid * V_DIM + v]);
        __syncthreads();
        for (int s = 128; s > 0; s >>= 1) {
            if (tid < s) red[tid] *= red[tid + s];
            __syncthreads();
        }
        if (tid == 0) {
            float soft = sgn_f(llr[v]) * red[0];
            out[v] = soft > 0.0f ? 1 : 0;
        }
        __syncthreads();
    }
}

extern "C" void kernel_launch(void* const* d_in, const int* in_sizes, int n_in,
                              void* d_out, int out_size, void* d_ws, size_t ws_size,
                              hipStream_t stream) {
    const float* llr      = (const float*)d_in[0];
    const int*   H        = (const int*)d_in[1];
    const int*   max_iter = (const int*)d_in[2];

    char* ws = (char*)d_ws;
    float*     m_vtoc = (float*)(ws);
    float*     m_ctov = (float*)(ws + (size_t)V_DIM * C_DIM * 4);
    _Float16*  HT     = (_Float16*)(ws + 2 * (size_t)V_DIM * C_DIM * 4);
    unsigned*  Hmask  = (unsigned*)(ws + 2 * (size_t)V_DIM * C_DIM * 4
                                       + (size_t)V_DIM * C_DIM * 2);

    ldpc_init_kernel<<<128, 256, 0, stream>>>(H, m_vtoc, m_ctov, HT, Hmask);
    ldpc_bp_kernel<<<1, 256, 0, stream>>>(llr, max_iter, m_vtoc, m_ctov,
                                          HT, Hmask, (int*)d_out);
}
